// MultiHeadAttention_79491254714583
// MI455X (gfx1250) — compile-verified, matrix operand order checked by reading, not run on hardware
//
#include <hip/hip_runtime.h>
#include <stdint.h>

#define B_  4
#define T_  2048
#define C_  1024
#define H_  16
#define D_  64
#define C3_ (3 * C_)

typedef __attribute__((ext_vector_type(16))) __bf16 v16bf;
typedef __attribute__((ext_vector_type(8)))  float  v8f;

union FragU { v16bf v; uint4 q[2]; unsigned short s[16]; };
union PackU { uint4 q; unsigned short s[8]; };

__device__ __forceinline__ unsigned short f2bf(float f) {
    union { float f; uint32_t u; } v; v.f = f;
    uint32_t u = v.u;
    uint32_t r = u + 0x7FFFu + ((u >> 16) & 1u);   // round-to-nearest-even
    return (unsigned short)(r >> 16);
}

__device__ __forceinline__ float exp2_fast(float x) {
    return __builtin_amdgcn_exp2f(x);              // v_exp_f32 (base-2 native)
}

// Two aligned 16B chunks -> one 16-element bf16 fragment (8 VGPRs).
__device__ __forceinline__ v16bf load_frag(const unsigned short* p_lo,
                                           const unsigned short* p_hi) {
    FragU f;
    f.q[0] = *(const uint4*)p_lo;
    f.q[1] = *(const uint4*)p_hi;
    return f.v;
}

__device__ __forceinline__ v8f wmma_bf16(v16bf a, v16bf b, v8f c) {
    // D(f32 16x16) = A(bf16 16x32) * B(bf16 32x16) + C
    return __builtin_amdgcn_wmma_f32_16x16x32_bf16(
        false, a, false, b, (short)0, c, false, false);
}

// CDNA5 async copy: 16B per lane, global -> LDS, tracked by ASYNCcnt.
__device__ __forceinline__ void async_load_b128(unsigned short* lds_dst,
                                                const unsigned short* gsrc) {
    unsigned int lp = (unsigned int)(uintptr_t)lds_dst;   // low 32b = LDS offset
    asm volatile("global_load_async_to_lds_b128 %0, %1, off"
                 :: "v"(lp), "v"(gsrc) : "memory");
}

__device__ __forceinline__ void wait_async0() {
    asm volatile("s_wait_asynccnt 0x0" ::: "memory");
}

// Swap values between lane halves (lane ^= 16): ds_swizzle group-of-32,
// xor_mask=16, and_mask=0x1f  ->  offset = (16<<10)|0x1f = 0x401F.
__device__ __forceinline__ float xor16(float x) {
    return __int_as_float(__builtin_amdgcn_ds_swizzle(__float_as_int(x), 0x401F));
}

// ---------------------------------------------------------------------------
// fp32 -> bf16 conversion, 4 elements/thread, exact sizing (n % 1024 == 0)
// ---------------------------------------------------------------------------
__global__ __launch_bounds__(256)
void cvt_f32_bf16(const float* __restrict__ in, unsigned short* __restrict__ out) {
    size_t i = (size_t)blockIdx.x * blockDim.x + threadIdx.x;
    float4 f = ((const float4*)in)[i];
    uint2 o;
    o.x = (uint32_t)f2bf(f.x) | ((uint32_t)f2bf(f.y) << 16);
    o.y = (uint32_t)f2bf(f.z) | ((uint32_t)f2bf(f.w) << 16);
    ((uint2*)out)[i] = o;
}

// ---------------------------------------------------------------------------
// bf16 GEMM: out[M,N] = A[M,K] * B[K,N], f32 accumulate.
// Block tile 128x128, 8 waves as 2(M) x 4(N), wave tile 64x32 (4x2 WMMA).
// Double-buffered LDS; A staged via CDNA5 async-to-LDS (issued for tile i+1
// before computing tile i, so the copy hides behind the WMMAs); B staged
// transposed ([N][K], stride 40 halfs -> fragments are aligned ds_load_b128).
// ---------------------------------------------------------------------------
template <int OUT_BF16>
__global__ __launch_bounds__(256)
void gemm_bf16(const unsigned short* __restrict__ A,
               const unsigned short* __restrict__ Bm,
               void* __restrict__ outp,
               int M, int N, int K) {
    __shared__ __align__(16) unsigned short As[2][128 * 40];
    __shared__ __align__(16) unsigned short Bs[2][128 * 40];

    const int tid  = threadIdx.x;
    const int wave = tid >> 5, lane = tid & 31;
    const int lm   = lane & 15, hi = lane >> 4;
    const int wm   = wave >> 2, wn = wave & 3;
    const int m0   = blockIdx.y * 128, n0 = blockIdx.x * 128;
    const int ko   = hi ? 8 : 0;    // K-lane split of the 16-bit WMMA layout
    const int c0   = tid * 2;

    v8f zero = {};
    v8f acc[4][2];
#pragma unroll
    for (int i = 0; i < 4; i++)
#pragma unroll
        for (int j = 0; j < 2; j++) acc[i][j] = zero;

    auto stage = [&](int k0, int buf) {
#pragma unroll
        for (int c = c0; c < c0 + 2; c++) {          // A: 2 async b128/thread
            int row = c >> 2, kof = (c & 3) * 8;
            async_load_b128(&As[buf][row * 40 + kof],
                            A + (size_t)(m0 + row) * K + k0 + kof);
        }
#pragma unroll
        for (int c = c0; c < c0 + 2; c++) {          // B: transpose into [N][K]
            int kr = c >> 4, nof = (c & 15) * 8;
            uint4 d = *(const uint4*)(Bm + (size_t)(k0 + kr) * N + n0 + nof);
            const unsigned short* ps = (const unsigned short*)&d;
#pragma unroll
            for (int i = 0; i < 8; i++) Bs[buf][(nof + i) * 40 + kr] = ps[i];
        }
    };

    stage(0, 0);
    wait_async0();
    __syncthreads();

    const int NK = K / 32;
    for (int i = 0; i < NK; i++) {
        const int cur = i & 1;
        if (i + 1 < NK) stage((i + 1) * 32, cur ^ 1);   // prefetch next tile

        v16bf af[4], bfr[2];
#pragma unroll
        for (int a = 0; a < 4; a++) {
            const unsigned short* base = &As[cur][(wm * 64 + a * 16 + lm) * 40];
            af[a] = load_frag(base + ko, base + 16 + ko);
        }
#pragma unroll
        for (int j = 0; j < 2; j++) {
            const unsigned short* base = &Bs[cur][(wn * 32 + j * 16 + lm) * 40];
            bfr[j] = load_frag(base + ko, base + 16 + ko);
        }
#pragma unroll
        for (int a = 0; a < 4; a++)
#pragma unroll
            for (int j = 0; j < 2; j++)
                acc[a][j] = wmma_bf16(af[a], bfr[j], acc[a][j]);

        wait_async0();
        __syncthreads();
    }

    // C layout: VGPR r -> row r (+8 for upper half-wave), col = lane%16.
#pragma unroll
    for (int i = 0; i < 4; i++)
#pragma unroll
        for (int j = 0; j < 2; j++) {
            int colg = n0 + wn * 32 + j * 16 + lm;
#pragma unroll
            for (int r = 0; r < 8; r++) {
                int rowg = m0 + wm * 64 + i * 16 + r + hi * 8;
                float val = acc[i][j][r];
                if (OUT_BF16)
                    ((unsigned short*)outp)[(size_t)rowg * N + colg] = f2bf(val);
                else
                    ((float*)outp)[(size_t)rowg * N + colg] = val;
            }
        }
}

// ---------------------------------------------------------------------------
// Causal flash attention, transposed formulation:
//   S^T = K * Q^T   (C layout: lane = query column, 8 key rows / half-wave)
//   O^T = V^T * P^T (P^T's B-fragment == the S^T accumulator registers)
// K staged verbatim via async-to-LDS (shared by all 8 waves), V staged
// transposed; both double-buffered. Wave-uniform skip of fully-masked blocks,
// mask-free fast path for interior blocks, softmax in the exp2 domain.
// ---------------------------------------------------------------------------
__global__ __launch_bounds__(256)
void attn_kernel(const unsigned short* __restrict__ qkv,
                 unsigned short* __restrict__ y) {
    __shared__ __align__(16) unsigned short Ks[2][32 * 72];  // K tile [key][d]
    __shared__ __align__(16) unsigned short Vs[2][64 * 40];  // V^T tile [d][key]

    const int tid  = threadIdx.x;
    const int wave = tid >> 5, lane = tid & 31;
    const int lm   = lane & 15, hi = lane >> 4;
    const int b    = blockIdx.y / H_, h = blockIdx.y % H_;
    const int qb   = blockIdx.x * 128;
    const int qw   = qb + wave * 16;
    const int q    = qw + lm;                 // this lane's query row
    const size_t base_bt = (size_t)b * T_;
    const int ko = hi ? 8 : 0;
    // 1/sqrt(D) * log2(e): softmax runs in the exp2 domain.
    const float SCL = 0.18033688011112042f;

    // Q as B-fragments (lane -> query column lm), kept for whole loop.
    v16bf qf0, qf1;
    {
        const unsigned short* qrow = qkv + (base_bt + q) * C3_ + h * D_;
        qf0 = load_frag(qrow + ko,      qrow + 16 + ko);   // d 0..31
        qf1 = load_frag(qrow + 32 + ko, qrow + 48 + ko);   // d 32..63
    }

    float m = -1e30f, l = 0.0f;
    v8f zero = {};
    v8f o[4];
#pragma unroll
    for (int t = 0; t < 4; t++) o[t] = zero;

    auto stageKV = [&](int j, int buf) {
        const int row = tid >> 3, cof = (tid & 7) * 8;
        // K rows verbatim: one async b128 per thread.
        async_load_b128(&Ks[buf][row * 72 + cof],
                        qkv + (base_bt + j + row) * C3_ + C_ + h * D_ + cof);
        // V transposed into [d][key].
        uint4 dv = *(const uint4*)(qkv + (base_bt + j + row) * C3_ +
                                   2 * C_ + h * D_ + cof);
        const unsigned short* ps = (const unsigned short*)&dv;
#pragma unroll
        for (int i = 0; i < 8; i++) Vs[buf][(cof + i) * 40 + row] = ps[i];
    };

    const int nblk = (qb + 128) / 32;          // causal extent of this block
    stageKV(0, 0);
    wait_async0();
    __syncthreads();

    for (int jb = 0; jb < nblk; jb++) {
        const int j   = jb * 32;
        const int cur = jb & 1;
        if (jb + 1 < nblk) stageKV(j + 32, cur ^ 1);   // prefetch next tile

        if (j <= qw + 15) {   // wave-uniform: skip fully-masked blocks
            // Load all 4 K fragments, then issue 4 independent WMMAs.
            v16bf ka[2][2];
#pragma unroll
            for (int t = 0; t < 2; t++)
#pragma unroll
                for (int s = 0; s < 2; s++) {
                    const unsigned short* kp =
                        &Ks[cur][(t * 16 + lm) * 72 + s * 32];
                    ka[t][s] = load_frag(kp + ko, kp + 16 + ko);
                }
            v8f s0 = zero, s1 = zero;
            s0 = wmma_bf16(ka[0][0], qf0, s0);     // keys j    .. j+15
            s1 = wmma_bf16(ka[1][0], qf0, s1);     // keys j+16 .. j+31
            s0 = wmma_bf16(ka[0][1], qf1, s0);
            s1 = wmma_bf16(ka[1][1], qf1, s1);

            // Scale (+ mask only on diagonal blocks) + lane-local max.
            float v0[8], v1[8];
            float mx = -1e30f;
            if (j + 31 > qw) {                     // diagonal: needs masking
#pragma unroll
                for (int r = 0; r < 8; r++) {
                    int key0 = j + r + hi * 8;
                    v0[r] = (key0      <= q) ? s0[r] * SCL : -1e30f;
                    v1[r] = (key0 + 16 <= q) ? s1[r] * SCL : -1e30f;
                    mx = fmaxf(mx, fmaxf(v0[r], v1[r]));
                }
            } else {                               // interior: mask-free
#pragma unroll
                for (int r = 0; r < 8; r++) {
                    v0[r] = s0[r] * SCL;
                    v1[r] = s1[r] * SCL;
                    mx = fmaxf(mx, fmaxf(v0[r], v1[r]));
                }
            }
            mx = fmaxf(mx, xor16(mx));             // combine key halves
            float mnew  = fmaxf(m, mx);
            float alpha = exp2_fast(m - mnew);
            m = mnew;

            // exp2 + pack P^T directly into the next WMMA's B fragment:
            // elem e<8 <-> key (hi?8:0)+e (tile0), e>=8 <-> 16+(hi?8:0)+e-8.
            float sum = 0.0f;
            FragU pu;
#pragma unroll
            for (int r = 0; r < 8; r++) {
                float p0 = exp2_fast(v0[r] - m);
                float p1 = exp2_fast(v1[r] - m);
                sum += p0 + p1;
                pu.s[r]     = f2bf(p0);
                pu.s[8 + r] = f2bf(p1);
            }
            sum += xor16(sum);
            l = l * alpha + sum;

#pragma unroll
            for (int t = 0; t < 4; t++)
#pragma unroll
                for (int r = 0; r < 8; r++) o[t][r] *= alpha;

            // Load all 4 V^T fragments, then 4 independent WMMAs.
            v16bf va[4];
#pragma unroll
            for (int t = 0; t < 4; t++) {
                const unsigned short* vr = &Vs[cur][(t * 16 + lm) * 40];
                va[t] = load_frag(vr + ko, vr + 16 + ko);
            }
#pragma unroll
            for (int t = 0; t < 4; t++)
                o[t] = wmma_bf16(va[t], pu.v, o[t]);   // O^T += V^T * P^T
        }

        wait_async0();
        __syncthreads();
    }

    // Epilogue: lane owns y row q, 8 consecutive d per tile -> b128 stores.
    float inv = 1.0f / l;
    unsigned short* yrow = y + (base_bt + q) * C_ + h * D_ + hi * 8;
#pragma unroll
    for (int t = 0; t < 4; t++) {
        PackU ou;
#pragma unroll
        for (int r = 0; r < 8; r++) ou.s[r] = f2bf(o[t][r] * inv);
        *(uint4*)(yrow + t * 16) = ou.q;
    }
}

// ---------------------------------------------------------------------------
extern "C" void kernel_launch(void* const* d_in, const int* in_sizes, int n_in,
                              void* d_out, int out_size, void* d_ws, size_t ws_size,
                              hipStream_t stream) {
    (void)in_sizes; (void)n_in; (void)out_size; (void)ws_size;
    const float* x      = (const float*)d_in[0];   // [B,T,C]
    const float* w_attn = (const float*)d_in[1];   // [C,3C]
    const float* w_proj = (const float*)d_in[2];   // [C,C]
    float* out = (float*)d_out;                    // [B,T,C]

    char* ws = (char*)d_ws;
    unsigned short* xb   = (unsigned short*)(ws);                         // 16 MB
    unsigned short* wab  = (unsigned short*)(ws + 16777216);              //  6 MB
    unsigned short* wpb  = (unsigned short*)(ws + 23068672);              //  2 MB
    unsigned short* qkvb = (unsigned short*)(ws + 25165824);              // 48 MB
    unsigned short* yb   = (unsigned short*)(ws + 75497472);              // 16 MB

    const int n_x  = B_ * T_ * C_;       // 8388608
    const int n_wa = C_ * C3_;           // 3145728
    const int n_wp = C_ * C_;            // 1048576

    cvt_f32_bf16<<<n_x  / 1024, 256, 0, stream>>>(x, xb);
    cvt_f32_bf16<<<n_wa / 1024, 256, 0, stream>>>(w_attn, wab);
    cvt_f32_bf16<<<n_wp / 1024, 256, 0, stream>>>(w_proj, wpb);

    // qkv = x @ w_attn   (M=8192, N=3072, K=1024), bf16 out
    gemm_bf16<1><<<dim3(C3_ / 128, (B_ * T_) / 128), 256, 0, stream>>>(
        xb, wab, (void*)qkvb, B_ * T_, C3_, C_);

    // flash attention -> y bf16
    attn_kernel<<<dim3(T_ / 128, B_ * H_), 256, 0, stream>>>(qkvb, yb);

    // out = y @ w_proj   (M=8192, N=1024, K=1024), f32 out
    gemm_bf16<0><<<dim3(C_ / 128, (B_ * T_) / 128), 256, 0, stream>>>(
        yb, wpb, (void*)out, B_ * T_, C_, C_);
}
